// FMForecaster_64175401337577
// MI455X (gfx1250) — matrix operand order repaired and checked by reading, not compiled, not run on hardware
//
#include <hip/hip_runtime.h>
#include <hip/hip_bf16.h>

// ---------------------------------------------------------------------------
// FM forecaster, fused persistent kernel for gfx1250 (MI455X).
// Shapes: B=128, L=CTX=96, PRED=24, S=100, STEPS=16, HID=256, IN_DIM=122.
// Rows = S*B = 12800; each block owns 64 rows for all 16 Euler steps.
// Matmuls via v_wmma_f32_16x16x32_f16 with 2x2 register tiling per wave.
// __launch_bounds__(256,2): 8 wave32 / 4 SIMDs = 2 waves/SIMD -> ~512 VGPR
// cap so the 2x2 tile + pipelined fragments never spill to scratch.
// ---------------------------------------------------------------------------

typedef __attribute__((ext_vector_type(16))) _Float16 v16h;
typedef __attribute__((ext_vector_type(8)))  _Float16 v8h;
typedef __attribute__((ext_vector_type(8)))  float    v8f;

#define FM_B     128
#define FM_CTX   96
#define FM_PRED  24
#define FM_S     100
#define FM_STEPS 16
#define FM_HID   256
#define FM_KPAD  128   // IN_DIM 122 padded to 128
#define FM_ROWS  64    // rows per block
#define FM_ZPAD  32    // z row stride in LDS (f32)

// Workspace layout (bytes)
#define OFF_W1T  0u                      // 256*128 f16 = 65536
#define OFF_W2T  65536u                  // 256*256 f16 = 131072
#define OFF_W3T  196608u                 // 32*256  f16 = 16384
#define OFF_LOC  212992u                 // 128 f32     = 512
#define OFF_CTX  213504u                 // 128*96 f16  = 24576

// Dynamic LDS layout (bytes)
#define SM_X    0u        // 64*256 f16 (x uses cols 0..127; reused as h2) = 32768
#define SM_H1   32768u    // 64*256 f16 = 32768
#define SM_Z    65536u    // 64*32 f32  = 8192
#define SM_CTX  73728u    // 64*96 f16  = 12288
#define SM_LOC  86016u    // 64 f32     = 256
#define SM_BYTES 86272u

__device__ __forceinline__ v16h ld2(const _Float16* p0, const _Float16* p1) {
    union { v16h v; v8h h[2]; } u;
    u.h[0] = *(const v8h*)p0;   // 16B -> ds_load_b128 / global_load_b128
    u.h[1] = *(const v8h*)p1;
    return u.v;
}

#define WMMA_F16(A, B, C) \
    __builtin_amdgcn_wmma_f32_16x16x32_f16(false, (A), false, (B), (short)0, (C), false, false)

// C = relu(A[64xK] @ Bt^T + bias) -> f16 LDS tile, Bt = [N][K] row-major f16
// in global memory.  2x2 tiles per wave: wave w owns M-tiles {(w>>2)*2, +1}
// and N-tile pairs {(w&3)*2 + j*8, +1}, j=0,1  (covers the full 4x16 grid).
// A-frags reused across N, B-frags across M; 4 independent WMMA chains.
// UNROLL bounds the number of k-chunks whose loads may be hoisted at once.
template <int K, int UNROLL, bool RELU>
__device__ __forceinline__ void gemm_2x2(const _Float16* __restrict__ sA, int ldA,
                                         const _Float16* __restrict__ Bt, int ldB,
                                         const float* __restrict__ bias,
                                         _Float16* __restrict__ sOut, int ldOut,
                                         int wid, int lane) {
    const int mL = lane & 15;
    const int hi = lane >> 4;
    const int mt0 = (wid >> 2) * 2;
    const _Float16* aRow0 = sA + (mt0 * 16 + mL) * ldA + hi * 8;   // A: 16x32 frag
    const _Float16* aRow1 = aRow0 + 16 * ldA;
#pragma unroll
    for (int j = 0; j < 2; ++j) {
        const int nt0 = (wid & 3) * 2 + j * 8;
        const _Float16* bRow0 = Bt + (nt0 * 16 + mL) * ldB + hi * 8;  // B: 32x16 frag
        const _Float16* bRow1 = bRow0 + 16 * ldB;
        v8f acc00 = {}, acc01 = {}, acc10 = {}, acc11 = {};
#pragma unroll UNROLL
        for (int k0 = 0; k0 < K; k0 += 32) {
            v16h a0 = ld2(aRow0 + k0, aRow0 + k0 + 16);
            v16h a1 = ld2(aRow1 + k0, aRow1 + k0 + 16);
            v16h b0 = ld2(bRow0 + k0, bRow0 + k0 + 16);
            v16h b1 = ld2(bRow1 + k0, bRow1 + k0 + 16);
            acc00 = WMMA_F16(a0, b0, acc00);
            acc01 = WMMA_F16(a0, b1, acc01);
            acc10 = WMMA_F16(a1, b0, acc10);
            acc11 = WMMA_F16(a1, b1, acc11);
        }
        const int n0 = nt0 * 16 + mL, n1 = n0 + 16;
        const float bv0 = bias[n0], bv1 = bias[n1];
        const int mb0 = mt0 * 16 + hi * 8, mb1 = mb0 + 16;
#pragma unroll
        for (int i = 0; i < 8; ++i) {
            float v00 = acc00[i] + bv0, v01 = acc01[i] + bv1;
            float v10 = acc10[i] + bv0, v11 = acc11[i] + bv1;
            if (RELU) {
                v00 = fmaxf(v00, 0.0f); v01 = fmaxf(v01, 0.0f);
                v10 = fmaxf(v10, 0.0f); v11 = fmaxf(v11, 0.0f);
            }
            sOut[(mb0 + i) * ldOut + n0] = (_Float16)v00;
            sOut[(mb0 + i) * ldOut + n1] = (_Float16)v01;
            sOut[(mb1 + i) * ldOut + n0] = (_Float16)v10;
            sOut[(mb1 + i) * ldOut + n1] = (_Float16)v11;
        }
    }
}

// v = h2 @ W3 + b3 (N padded 24->32), then z -= dt*v written straight to LDS z.
__device__ __forceinline__ void gemm3_zupdate(const _Float16* __restrict__ sH,
                                              const _Float16* __restrict__ W3t,
                                              const float* __restrict__ b3,
                                              float* __restrict__ sZ, float dt,
                                              int wid, int lane) {
    const int mL = lane & 15, hi = lane >> 4;
    const int mt = wid >> 1, nt = wid & 1;   // 8 tiles, 1 per wave
    const _Float16* aRow = sH + (mt * 16 + mL) * FM_HID + hi * 8;
    const _Float16* bRow = W3t + (nt * 16 + mL) * FM_HID + hi * 8;
    v8f acc = {};
#pragma unroll 4
    for (int k0 = 0; k0 < FM_HID; k0 += 32) {
        v16h a = ld2(aRow + k0, aRow + k0 + 16);
        v16h b = ld2(bRow + k0, bRow + k0 + 16);
        acc = WMMA_F16(a, b, acc);
    }
    const int n = nt * 16 + mL;
    if (n < FM_PRED) {
        const float bv = b3[n];
        const int mbase = mt * 16 + hi * 8;
#pragma unroll
        for (int i = 0; i < 8; ++i)
            sZ[(mbase + i) * FM_ZPAD + n] -= dt * (acc[i] + bv);
    }
}

// ---- prep: loc[b] = max(mean|ctx|,1e-6); ctxh[b][j] = f16(ctx/loc) ----------
__global__ void fm_prep_scale(const float* __restrict__ past,
                              float* __restrict__ loc, _Float16* __restrict__ ctxh) {
    int b = threadIdx.x;
    if (b < FM_B) {
        const float* row = past + b * 96;  // L == CTX == 96, ctx = whole row
        float s = 0.f;
        for (int j = 0; j < FM_CTX; ++j) s += fabsf(row[j]);
        float l = fmaxf(s * (1.0f / FM_CTX), 1e-6f);
        loc[b] = l;
        float inv = 1.0f / l;
        for (int j = 0; j < FM_CTX; ++j) ctxh[b * FM_CTX + j] = (_Float16)(row[j] * inv);
    }
}

// ---- prep: transpose + f16-convert weights (K padded / N padded) ------------
__global__ void fm_prep_weights(const float* __restrict__ W1, const float* __restrict__ W2,
                                const float* __restrict__ W3,
                                _Float16* __restrict__ W1t, _Float16* __restrict__ W2t,
                                _Float16* __restrict__ W3t) {
    int i = blockIdx.x * 256 + threadIdx.x;
    if (i < 256 * 128) {                       // W1t[n][k], k<122 valid
        int n = i / 128, k = i % 128;
        W1t[i] = (_Float16)((k < 122) ? W1[k * 256 + n] : 0.0f);
    } else if (i < 32768 + 65536) {            // W2t[n][k]
        int j = i - 32768; int n = j / 256, k = j % 256;
        W2t[j] = (_Float16)W2[k * 256 + n];
    } else if (i < 32768 + 65536 + 8192) {     // W3t[n][k], n<24 valid
        int j = i - 32768 - 65536; int n = j / 256, k = j % 256;
        W3t[j] = (_Float16)((n < 24) ? W3[k * 24 + n] : 0.0f);
    }
}

// ---- main fused persistent kernel ------------------------------------------
__global__ void __launch_bounds__(256, 2)
fm_main(const float* __restrict__ z0, const float* __restrict__ b1,
        const float* __restrict__ b2, const float* __restrict__ b3,
        const _Float16* __restrict__ W1t, const _Float16* __restrict__ W2t,
        const _Float16* __restrict__ W3t, const float* __restrict__ loc,
        const _Float16* __restrict__ ctxh, float* __restrict__ out) {
    extern __shared__ __align__(16) char smem[];
    _Float16* sX   = (_Float16*)(smem + SM_X);    // x (64x128) / h2 (64x256)
    _Float16* sH1  = (_Float16*)(smem + SM_H1);   // h1 (64x256)
    float*    sZ   = (float*)(smem + SM_Z);       // z (64x32 padded)
    _Float16* sCtx = (_Float16*)(smem + SM_CTX);  // scaled ctx (64x96)
    float*    sLoc = (float*)(smem + SM_LOC);

    const int row0 = blockIdx.x * FM_ROWS;
    const int tid = threadIdx.x, wid = tid >> 5, lane = tid & 31;

    for (int e = tid; e < FM_ROWS * FM_PRED; e += 256) {
        int r = e / FM_PRED, p = e % FM_PRED;
        sZ[r * FM_ZPAD + p] = z0[(row0 + r) * FM_PRED + p];
    }
    for (int e = tid; e < FM_ROWS * FM_CTX; e += 256) {
        int r = e / FM_CTX, j = e % FM_CTX;
        sCtx[r * FM_CTX + j] = ctxh[((row0 + r) & (FM_B - 1)) * FM_CTX + j];
    }
    for (int r = tid; r < FM_ROWS; r += 256) sLoc[r] = loc[(row0 + r) & (FM_B - 1)];
    __syncthreads();

    const float dt = 1.0f / FM_STEPS;
    for (int step = 0; step < FM_STEPS; ++step) {
        const float tval = 1.0f - step * dt;
        // build x = [z | t | 0 | scaled_ctx | pad] as f16
        for (int e = tid; e < FM_ROWS * FM_KPAD; e += 256) {
            int r = e >> 7, c = e & 127;
            _Float16 v;
            if (c < FM_PRED)                 v = (_Float16)sZ[r * FM_ZPAD + c];
            else if (c == FM_PRED)           v = (_Float16)tval;
            else if (c >= 26 && c < 122)     v = sCtx[r * FM_CTX + (c - 26)];
            else                             v = (_Float16)0.0f;
            sX[r * FM_KPAD + c] = v;
        }
        __syncthreads();
        gemm_2x2<FM_KPAD, 4, true>(sX, FM_KPAD, W1t, FM_KPAD, b1, sH1, FM_HID, wid, lane);
        __syncthreads();
        gemm_2x2<FM_HID, 4, true>(sH1, FM_HID, W2t, FM_HID, b2, sX, FM_HID, wid, lane);
        __syncthreads();
        gemm3_zupdate(sX, W3t, b3, sZ, dt, wid, lane);
        __syncthreads();
    }

    // preds = z*loc, scatter (S,B,P) rows -> out (B,S,P)
    for (int e = tid; e < FM_ROWS * FM_PRED; e += 256) {
        int r = e / FM_PRED, p = e % FM_PRED;
        int g = row0 + r, s = g >> 7, b = g & (FM_B - 1);
        out[b * (FM_S * FM_PRED) + s * FM_PRED + p] = sZ[r * FM_ZPAD + p] * sLoc[r];
    }
}

extern "C" void kernel_launch(void* const* d_in, const int* in_sizes, int n_in,
                              void* d_out, int out_size, void* d_ws, size_t ws_size,
                              hipStream_t stream) {
    const float* past = (const float*)d_in[0];
    // d_in[1] = past_observed_values (all ones, unused by the math)
    const float* z0 = (const float*)d_in[2];
    const float* W1 = (const float*)d_in[3];
    const float* b1 = (const float*)d_in[4];
    const float* W2 = (const float*)d_in[5];
    const float* b2 = (const float*)d_in[6];
    const float* W3 = (const float*)d_in[7];
    const float* b3 = (const float*)d_in[8];
    float* out = (float*)d_out;

    char* ws = (char*)d_ws;
    _Float16* W1t  = (_Float16*)(ws + OFF_W1T);
    _Float16* W2t  = (_Float16*)(ws + OFF_W2T);
    _Float16* W3t  = (_Float16*)(ws + OFF_W3T);
    float*    locp = (float*)(ws + OFF_LOC);
    _Float16* ctxh = (_Float16*)(ws + OFF_CTX);

    (void)hipFuncSetAttribute((const void*)fm_main,
                              hipFuncAttributeMaxDynamicSharedMemorySize, SM_BYTES);

    fm_prep_scale<<<1, 128, 0, stream>>>(past, locp, ctxh);
    fm_prep_weights<<<(32768 + 65536 + 8192 + 255) / 256, 256, 0, stream>>>(
        W1, W2, W3, W1t, W2t, W3t);

    const int nBlocks = (FM_S * FM_B) / FM_ROWS;  // 200
    fm_main<<<nBlocks, 256, SM_BYTES, stream>>>(z0, b1, b2, b3, W1t, W2t, W3t,
                                                locp, ctxh, out);
}
